// Head_25391846654747
// MI455X (gfx1250) — compile-verified, matrix operand order checked by reading, not run on hardware
//
#include <hip/hip_runtime.h>
#include <stdint.h>

#define B_ 8
#define T_ 2048
#define C_ 1024
#define H_ 64

typedef unsigned short u16t;
typedef __attribute__((ext_vector_type(16))) __bf16 v16bf;
typedef __attribute__((ext_vector_type(8)))  __bf16 v8bf;
typedef __attribute__((ext_vector_type(16))) float  v16f;
typedef __attribute__((ext_vector_type(8)))  float  v8f;

union U16 {
  v16bf v;
  u16t  u[16];
  uint4 q[2];
};
union U8 {
  v8bf v;
  u16t u[8];
  uint4 q;
};

// Native bf16 convert (RNE) — lets clang emit v_cvt_pk_bf16_f32 instead of
// bit-twiddled rounding (3 VALU/elem -> 0.5 VALU/elem).
__device__ __forceinline__ u16t f2bf(float f) {
  __bf16 h = (__bf16)f;
  return __builtin_bit_cast(u16t, h);
}
__device__ __forceinline__ v8bf cvt8(v8f x) {
  return __builtin_convertvector(x, v8bf);
}

__device__ __forceinline__ v8f wmma_bf16(v16bf a, v16bf b, v8f c) {
  // v_wmma_f32_16x16x32_bf16: D = A(16x32) * B(32x16) + C(16x16 f32)
  return __builtin_amdgcn_wmma_f32_16x16x32_bf16(false, a, false, b, (short)0, c,
                                                 false, false);
}

// A/B operand loader from a row-major bf16 matrix (A: lane=row of M; for B the
// "rows" are the columns of the 32x16 operand, i.e. base points at Bt[n][k]).
// ISA layout (16-bit A 16x32): lanes 0-15 -> K 0..7 & 16..23, lanes 16-31 -> K 8..15 & 24..31.
__device__ __forceinline__ v16bf load_bf_tile(const u16t* __restrict__ base, int ld, int lane) {
  const int row  = lane & 15;
  const int koff = (lane >> 4) << 3;
  const u16t* p = base + row * ld + koff;
  U16 t;
  t.q[0] = *(const uint4*)(p);        // K = koff .. koff+7
  t.q[1] = *(const uint4*)(p + 16);   // K = koff+16 .. koff+23
  return t.v;
}

// Same A layout but sourced from fp32 memory, converting to bf16 inline.
__device__ __forceinline__ v16bf load_f32_tile_bf(const float* __restrict__ base, int ld, int lane) {
  const int row  = lane & 15;
  const int koff = (lane >> 4) << 3;
  const float* p = base + row * ld + koff;
  float4 f0 = *(const float4*)(p);
  float4 f1 = *(const float4*)(p + 4);
  float4 f2 = *(const float4*)(p + 16);
  float4 f3 = *(const float4*)(p + 20);
  v16f f;
  f[0]  = f0.x; f[1]  = f0.y; f[2]  = f0.z; f[3]  = f0.w;
  f[4]  = f1.x; f[5]  = f1.y; f[6]  = f1.z; f[7]  = f1.w;
  f[8]  = f2.x; f[9]  = f2.y; f[10] = f2.z; f[11] = f2.w;
  f[12] = f3.x; f[13] = f3.y; f[14] = f3.z; f[15] = f3.w;
  return __builtin_convertvector(f, v16bf);   // packed v_cvt_pk_bf16_f32
}

// Row reductions across the 16 lanes sharing a C/D-layout row (xor<=8 stays in half).
__device__ __forceinline__ float rowmax16(float v) {
  v = fmaxf(v, __shfl_xor(v, 1, 32));
  v = fmaxf(v, __shfl_xor(v, 2, 32));
  v = fmaxf(v, __shfl_xor(v, 4, 32));
  v = fmaxf(v, __shfl_xor(v, 8, 32));
  return v;
}
__device__ __forceinline__ float rowsum16(float v) {
  v += __shfl_xor(v, 1, 32);
  v += __shfl_xor(v, 2, 32);
  v += __shfl_xor(v, 4, 32);
  v += __shfl_xor(v, 8, 32);
  return v;
}

// ---------------------------------------------------------------------------
// Kernel 1: weights fp32 [C,H] -> bf16 transposed [3][H][C]  (slot 0=Wq,1=Wk,2=Wv)
// ---------------------------------------------------------------------------
__global__ __launch_bounds__(256) void wt_conv(const float* __restrict__ Wq,
                                               const float* __restrict__ Wk,
                                               const float* __restrict__ Wv,
                                               u16t* __restrict__ Wt) {
  int id = blockIdx.x * blockDim.x + threadIdx.x;
  if (id >= 3 * H_ * C_) return;
  int slot = id / (H_ * C_);
  int rem  = id - slot * (H_ * C_);
  int h = rem / C_;
  int c = rem - h * C_;
  const float* W = (slot == 0) ? Wq : (slot == 1) ? Wk : Wv;
  Wt[id] = f2bf(W[c * H_ + h]);
}

// ---------------------------------------------------------------------------
// Kernel 2: QKV projection.  One wave per 16x64 output tile.
//   q,k stored row-major bf16 [B*T, 64];  v stored transposed bf16 [B, 64, T].
// ---------------------------------------------------------------------------
__global__ __launch_bounds__(256) void qkv_proj(const float* __restrict__ x,
                                                const u16t* __restrict__ Wt,
                                                u16t* __restrict__ qb,
                                                u16t* __restrict__ kb,
                                                u16t* __restrict__ vt) {
  const int lane = threadIdx.x & 31;
  const int wid  = threadIdx.x >> 5;
  const int tile = blockIdx.x * 8 + wid;          // 0 .. 3071
  const int mat  = tile >> 10;                    // /1024 : 0=Q,1=K,2=V
  const int rt   = tile & 1023;                   // row tile of flat [B*T, C]
  const int row0 = rt << 4;

  const u16t* wt = Wt + mat * (H_ * C_);
  const float* xr = x + (size_t)row0 * C_;

  v8f acc0 = {0.f,0.f,0.f,0.f,0.f,0.f,0.f,0.f};
  v8f acc1 = acc0, acc2 = acc0, acc3 = acc0;

  for (int k0 = 0; k0 < C_; k0 += 32) {
    v16bf a = load_f32_tile_bf(xr + k0, C_, lane);
    v16bf b0 = load_bf_tile(wt +  0 * C_ + k0, C_, lane);
    v16bf b1 = load_bf_tile(wt + 16 * C_ + k0, C_, lane);
    v16bf b2 = load_bf_tile(wt + 32 * C_ + k0, C_, lane);
    v16bf b3 = load_bf_tile(wt + 48 * C_ + k0, C_, lane);
    acc0 = wmma_bf16(a, b0, acc0);
    acc1 = wmma_bf16(a, b1, acc1);
    acc2 = wmma_bf16(a, b2, acc2);
    acc3 = wmma_bf16(a, b3, acc3);
  }

  const int nl = lane & 15, hi = lane >> 4;
  v8f acc[4] = {acc0, acc1, acc2, acc3};
  if (mat < 2) {
    u16t* out = (mat == 0) ? qb : kb;
    #pragma unroll
    for (int n = 0; n < 4; ++n) {
      U8 pk; pk.v = cvt8(acc[n]);
      #pragma unroll
      for (int r = 0; r < 8; ++r)
        out[(size_t)(row0 + r + 8 * hi) * H_ + n * 16 + nl] = pk.u[r];
    }
  } else {
    const int bidx = row0 >> 11;          // batch
    const int t0   = row0 & (T_ - 1);     // 16-aligned, never crosses batch
    #pragma unroll
    for (int n = 0; n < 4; ++n) {
      const int h = n * 16 + nl;
      U8 pk; pk.v = cvt8(acc[n]);
      *(uint4*)(vt + (size_t)bidx * H_ * T_ + (size_t)h * T_ + t0 + 8 * hi) = pk.q;
    }
  }
}

// ---------------------------------------------------------------------------
// Kernel 3: flash attention.  One wave per 16-query tile; 32-key blocks.
// ---------------------------------------------------------------------------
__global__ __launch_bounds__(128) void attn(const u16t* __restrict__ qbuf,
                                            const u16t* __restrict__ kbuf,
                                            const u16t* __restrict__ vtbuf,
                                            float* __restrict__ out) {
  __shared__ u16t pbuf[4 * 16 * 40];              // per-wave 16x32 bf16 (stride 40)
  const int lane = threadIdx.x & 31;
  const int wid  = threadIdx.x >> 5;
  const int nl = lane & 15, hi = lane >> 4;

  const int tile = blockIdx.x * 4 + wid;          // 0 .. 1023
  const int bidx = tile >> 7;                     // 128 tiles per batch
  const int q0   = (tile & 127) << 4;             // query row within batch

  const u16t* qp = qbuf  + ((size_t)bidx * T_ + q0) * H_;
  const u16t* kp = kbuf  + (size_t)bidx * T_ * H_;
  const u16t* vp = vtbuf + (size_t)bidx * H_ * T_;
  u16t* pb = pbuf + wid * 16 * 40;

  // Q tile resident in A layout (16 rows x 64, two K=32 chunks)
  v16bf qa0 = load_bf_tile(qp +  0, H_, lane);
  v16bf qa1 = load_bf_tile(qp + 32, H_, lane);

  v8f z = {0.f,0.f,0.f,0.f,0.f,0.f,0.f,0.f};
  v8f o0 = z, o1 = z, o2 = z, o3 = z;
  float mr[8], lr[8];
  #pragma unroll
  for (int r = 0; r < 8; ++r) { mr[r] = -1e30f; lr[r] = 0.f; }

  const float SC = 0.125f * 1.44269504088896340736f;   // scale * log2(e)
  const int kend = q0 + 16;                            // exclusive (causal)

  for (int kb = 0; kb < kend; kb += 32) {
    if (kb + 32 < kend)                                // prefetch next K tile
      __builtin_prefetch((const char*)(kp + (size_t)(kb + 32) * H_) + lane * 128, 0, 0);

    // S = Q Kt : two 16x16 tiles (keys kb..kb+15 and kb+16..kb+31)
    const u16t* kt = kp + (size_t)kb * H_;
    v8f s0 = z, s1 = z;
    v16bf bk;
    bk = load_bf_tile(kt + 0,            H_, lane); s0 = wmma_bf16(qa0, bk, s0);
    bk = load_bf_tile(kt + 32,           H_, lane); s0 = wmma_bf16(qa1, bk, s0);
    bk = load_bf_tile(kt + 16 * H_ + 0,  H_, lane); s1 = wmma_bf16(qa0, bk, s1);
    bk = load_bf_tile(kt + 16 * H_ + 32, H_, lane); s1 = wmma_bf16(qa1, bk, s1);

    // online softmax (rows = r + 8*hi, cols = nl / 16+nl)
    v8f p0, p1;
    #pragma unroll
    for (int r = 0; r < 8; ++r) {
      const int qi = q0 + r + 8 * hi;
      float a0 = (kb + nl      <= qi) ? s0[r] * SC : -1e30f;
      float a1 = (kb + 16 + nl <= qi) ? s1[r] * SC : -1e30f;
      float rm = rowmax16(fmaxf(a0, a1));
      float mn = fmaxf(mr[r], rm);
      float corr = exp2f(mr[r] - mn);
      mr[r] = mn;
      p0[r] = exp2f(a0 - mn);
      p1[r] = exp2f(a1 - mn);
      lr[r] = lr[r] * corr + rowsum16(p0[r] + p1[r]);
      o0[r] *= corr; o1[r] *= corr; o2[r] *= corr; o3[r] *= corr;
    }

    // P: D layout -> LDS -> A layout (same-wave DS ops are in-order)
    U8 c0, c1;
    c0.v = cvt8(p0);
    c1.v = cvt8(p1);
    #pragma unroll
    for (int r = 0; r < 8; ++r) {
      const int m = r + 8 * hi;
      pb[m * 40 + nl]      = c0.u[r];
      pb[m * 40 + 16 + nl] = c1.u[r];
    }
    U16 tp;
    const u16t* pp = pb + nl * 40 + (hi << 3);
    tp.q[0] = *(const uint4*)(pp);
    tp.q[1] = *(const uint4*)(pp + 16);
    v16bf pa = tp.v;

    // O += P (16x32) * V (32x16 per H chunk), B from transposed V [64,T]
    v16bf bv;
    bv = load_bf_tile(vp +  0 * T_ + kb, T_, lane); o0 = wmma_bf16(pa, bv, o0);
    bv = load_bf_tile(vp + 16 * T_ + kb, T_, lane); o1 = wmma_bf16(pa, bv, o1);
    bv = load_bf_tile(vp + 32 * T_ + kb, T_, lane); o2 = wmma_bf16(pa, bv, o2);
    bv = load_bf_tile(vp + 48 * T_ + kb, T_, lane); o3 = wmma_bf16(pa, bv, o3);
  }

  // epilogue: normalize and store fp32 [B*T, 64]
  float* op = out + ((size_t)bidx * T_ + q0) * H_;
  v8f oo[4] = {o0, o1, o2, o3};
  #pragma unroll
  for (int n = 0; n < 4; ++n) {
    #pragma unroll
    for (int r = 0; r < 8; ++r)
      op[(size_t)(r + 8 * hi) * H_ + n * 16 + nl] = oo[n][r] / lr[r];
  }
}

// ---------------------------------------------------------------------------
extern "C" void kernel_launch(void* const* d_in, const int* in_sizes, int n_in,
                              void* d_out, int out_size, void* d_ws, size_t ws_size,
                              hipStream_t stream) {
  const float* x  = (const float*)d_in[0];
  const float* Wk = (const float*)d_in[1];
  const float* Wq = (const float*)d_in[2];
  const float* Wv = (const float*)d_in[3];
  float* out = (float*)d_out;

  // workspace layout (bf16): Wt[3*64*1024] | q[B*T*64] | k[B*T*64] | vT[B*64*T]
  u16t* Wt = (u16t*)d_ws;
  u16t* qb = Wt + 3 * H_ * C_;
  u16t* kb = qb + B_ * T_ * H_;
  u16t* vt = kb + B_ * T_ * H_;

  const int wn = 3 * H_ * C_;
  wt_conv<<<(wn + 255) / 256, 256, 0, stream>>>(Wq, Wk, Wv, Wt);

  // 3 mats * (B*T/16) tiles = 3072 waves, 8 waves/block
  qkv_proj<<<384, 256, 0, stream>>>(x, Wt, qb, kb, vt);

  // B*T/16 = 1024 query tiles, 4 waves/block
  attn<<<256, 128, 0, stream>>>(qb, kb, vt, out);
}
